// GAT_62878321213786
// MI455X (gfx1250) — compile-verified, matrix operand order checked by reading, not run on hardware
//
#include <hip/hip_runtime.h>
#include <hip/hip_bf16.h>

// ---------------------------------------------------------------------------
// Types for CDNA5 WMMA (wave32): v_wmma_f32_16x16x32_bf16
// ---------------------------------------------------------------------------
typedef __attribute__((ext_vector_type(16))) __bf16          v16bf;
typedef __attribute__((ext_vector_type(16))) unsigned short  v16us;
typedef __attribute__((ext_vector_type(8)))  float           v8f;

__device__ __forceinline__ unsigned short f2bf(float f) {
    // round-to-nearest-even f32 -> bf16
    unsigned int u = __float_as_uint(f);
    u += 0x7FFFu + ((u >> 16) & 1u);
    return (unsigned short)(u >> 16);
}

__device__ __forceinline__ unsigned int pk2(unsigned short lo, unsigned short hi) {
    return (unsigned int)lo | ((unsigned int)hi << 16);
}

// ---------------------------------------------------------------------------
// One-shot f32 -> packed bf16 conversion (8 elements / thread). n % 8 == 0.
// ---------------------------------------------------------------------------
__global__ void f32_to_bf16_kernel(const float* __restrict__ in,
                                   unsigned short* __restrict__ out, int n) {
    const int i = (blockIdx.x * blockDim.x + threadIdx.x) * 8;
    if (i >= n) return;
    const float4 a = *(const float4*)(in + i);
    const float4 b = *(const float4*)(in + i + 4);
    uint4 o;
    o.x = pk2(f2bf(a.x), f2bf(a.y));
    o.y = pk2(f2bf(a.z), f2bf(a.w));
    o.z = pk2(f2bf(b.x), f2bf(b.y));
    o.w = pk2(f2bf(b.z), f2bf(b.w));
    *(uint4*)(out + i) = o;
}

// ELU fused with f32 -> bf16 conversion (layer-2 input). n % 8 == 0.
__global__ void elu_to_bf16_kernel(const float* __restrict__ in,
                                   unsigned short* __restrict__ out, int n) {
    const int i = (blockIdx.x * blockDim.x + threadIdx.x) * 8;
    if (i >= n) return;
    float v[8];
    const float4 a = *(const float4*)(in + i);
    const float4 b = *(const float4*)(in + i + 4);
    v[0]=a.x; v[1]=a.y; v[2]=a.z; v[3]=a.w; v[4]=b.x; v[5]=b.y; v[6]=b.z; v[7]=b.w;
    unsigned short r[8];
    #pragma unroll
    for (int j = 0; j < 8; ++j) {
        const float e = v[j] > 0.f ? v[j] : __expf(v[j]) - 1.0f;
        r[j] = f2bf(e);
    }
    uint4 o;
    o.x = pk2(r[0], r[1]); o.y = pk2(r[2], r[3]);
    o.z = pk2(r[4], r[5]); o.w = pk2(r[6], r[7]);
    *(uint4*)(out + i) = o;
}

// ---------------------------------------------------------------------------
// Load a 16(row)x32(K) bf16 fragment directly from packed-bf16 memory.
// p = &Mat[row*K + kbase] (ushort units), row = lane&15, kbase = 8*(lane>>4)+k0.
// VGPRs 0-3 hold K = kbase+0..7, VGPRs 4-7 hold K = kbase+16..23.
// Two global_load_b128, no conversion ALU.
// ---------------------------------------------------------------------------
__device__ __forceinline__ v16bf load_frag_bf(const unsigned short* __restrict__ p) {
    union { uint4 q[2]; v16us v; } u;
    u.q[0] = *(const uint4*)(p);        // K offsets 0..7
    u.q[1] = *(const uint4*)(p + 16);   // K offsets 16..23
    return __builtin_bit_cast(v16bf, u.v);
}

// ---------------------------------------------------------------------------
// C[M,Nout] = A[M,K] * W[Nout,K]^T (+ bias[Nout]); A,W packed bf16, C f32.
// One wave computes a 16-row x 64-col strip (4 WMMA accumulators, A frag
// reused 4x). 8 waves/block, grid = (ceil(M/128), Nout/64).
// Requires M % 16 == 0, K % 32 == 0, Nout % 64 == 0.
// ---------------------------------------------------------------------------
__global__ void __launch_bounds__(256)
wmma_gemm_kernel(const unsigned short* __restrict__ A,
                 const unsigned short* __restrict__ W,
                 const float* __restrict__ bias, float* __restrict__ C,
                 int M, int K, int Nout) {
    const int lane = threadIdx.x & 31;
    const int wv   = threadIdx.x >> 5;
    const int rt   = blockIdx.x * 8 + wv;          // 16-row tile index
    if (rt * 16 >= M) return;
    const int nb   = blockIdx.y * 64;
    const int l15  = lane & 15;
    const int kh   = (lane >> 4) << 3;             // 0 or 8

    v8f acc[4];
    #pragma unroll
    for (int t = 0; t < 4; ++t)
        acc[t] = (v8f){0.f,0.f,0.f,0.f,0.f,0.f,0.f,0.f};

    const unsigned short* arow = A + (size_t)(rt * 16 + l15) * K + kh;
    const unsigned short* w0 = W + (size_t)(nb +  0 + l15) * K + kh;
    const unsigned short* w1 = W + (size_t)(nb + 16 + l15) * K + kh;
    const unsigned short* w2 = W + (size_t)(nb + 32 + l15) * K + kh;
    const unsigned short* w3 = W + (size_t)(nb + 48 + l15) * K + kh;

    for (int k0 = 0; k0 < K; k0 += 32) {
        const v16bf af = load_frag_bf(arow + k0);
        const v16bf b0 = load_frag_bf(w0 + k0);
        const v16bf b1 = load_frag_bf(w1 + k0);
        const v16bf b2 = load_frag_bf(w2 + k0);
        const v16bf b3 = load_frag_bf(w3 + k0);
        acc[0] = __builtin_amdgcn_wmma_f32_16x16x32_bf16(false, af, false, b0, (short)0, acc[0], false, false);
        acc[1] = __builtin_amdgcn_wmma_f32_16x16x32_bf16(false, af, false, b1, (short)0, acc[1], false, false);
        acc[2] = __builtin_amdgcn_wmma_f32_16x16x32_bf16(false, af, false, b2, (short)0, acc[2], false, false);
        acc[3] = __builtin_amdgcn_wmma_f32_16x16x32_bf16(false, af, false, b3, (short)0, acc[3], false, false);
    }

    // C/D layout: VGPR r -> row rt*16 + r + 8*(lane>>4), col nb + 16*t + (lane&15)
    const int row0 = rt * 16 + ((lane >> 4) << 3);
    #pragma unroll
    for (int t = 0; t < 4; ++t) {
        const int col = nb + t * 16 + l15;
        const float bv = bias ? bias[col] : 0.0f;
        #pragma unroll
        for (int r = 0; r < 8; ++r)
            C[(size_t)(row0 + r) * Nout + col] = acc[t][r] + bv;
    }
}

// ---------------------------------------------------------------------------
// el[n,h] = dot(feat[n,h,:], attn_l[h,:]) ; er likewise. One wave per (n,h).
// ---------------------------------------------------------------------------
template <int H>
__global__ void attn_dot_kernel(const float* __restrict__ feat,
                                const float* __restrict__ al,
                                const float* __restrict__ ar,
                                float* __restrict__ el, float* __restrict__ er,
                                int NH) {
    const int widx = blockIdx.x * (blockDim.x >> 5) + (threadIdx.x >> 5);
    if (widx >= NH) return;
    const int lane = threadIdx.x & 31;
    const int h = widx % H;
    const float* f = feat + (size_t)widx * 128;
    const float* L = al + h * 128;
    const float* R = ar + h * 128;
    float sl = 0.f, sr = 0.f;
    #pragma unroll
    for (int j = 0; j < 4; ++j) {
        const float v = f[lane + 32 * j];
        sl += v * L[lane + 32 * j];
        sr += v * R[lane + 32 * j];
    }
    #pragma unroll
    for (int off = 16; off > 0; off >>= 1) {
        sl += __shfl_xor(sl, off, 32);
        sr += __shfl_xor(sr, off, 32);
    }
    if (lane == 0) { el[widx] = sl; er[widx] = sr; }
}

__global__ void init_ms_kernel(float* __restrict__ m, float* __restrict__ s, int n) {
    const int i = blockIdx.x * blockDim.x + threadIdx.x;
    if (i < n) { m[i] = -3.0e38f; s[i] = 0.0f; }
}

__device__ __forceinline__ void atomicMaxFloat(float* addr, float v) {
    if (v >= 0.0f) atomicMax((int*)addr, __float_as_int(v));
    else           atomicMin((unsigned int*)addr, __float_as_uint(v));
}

// Pass 1: e = leaky_relu(el[src]+er[dst]); store e; segment max into m[dst].
template <int H>
__global__ void edge_max_kernel(const int* __restrict__ src, const int* __restrict__ dst,
                                const float* __restrict__ el, const float* __restrict__ er,
                                float* __restrict__ ebuf, float* __restrict__ m, int EH) {
    const int t = blockIdx.x * blockDim.x + threadIdx.x;
    if (t >= EH) return;
    const int e = t / H, h = t - e * H;
    const int s = src[e], d = dst[e];
    float x = el[s * H + h] + er[d * H + h];
    x = x > 0.0f ? x : 0.2f * x;
    ebuf[t] = x;
    atomicMaxFloat(&m[d * H + h], x);
}

// Pass 2: ex = exp(e - m[dst]); store ex; segment sum into s[dst].
template <int H>
__global__ void edge_exp_kernel(const int* __restrict__ dst, float* __restrict__ ebuf,
                                const float* __restrict__ m, float* __restrict__ ssum,
                                int EH) {
    const int t = blockIdx.x * blockDim.x + threadIdx.x;
    if (t >= EH) return;
    const int e = t / H, h = t - e * H;
    const int d = dst[e];
    const float ex = __expf(ebuf[t] - m[d * H + h]);
    ebuf[t] = ex;
    atomicAdd(&ssum[d * H + h], ex);
}

// Pass 3: out[dst,h,:] += (ex/s[dst,h]) * feat[src,h,:]. One wave per (e,h);
// each lane handles 4 contiguous floats of the 128-wide feature.
template <int H>
__global__ void __launch_bounds__(256)
edge_aggregate_kernel(const int* __restrict__ src, const int* __restrict__ dst,
                      const float* __restrict__ ebuf, const float* __restrict__ ssum,
                      const float* __restrict__ feat, float* __restrict__ out, int EH) {
    const int widx = blockIdx.x * (blockDim.x >> 5) + (threadIdx.x >> 5);
    if (widx >= EH) return;
    const int lane = threadIdx.x & 31;
    const int e = widx / H, h = widx - e * H;
    const int s = src[e], d = dst[e];
    const float alpha = ebuf[widx] / ssum[d * H + h];
    const float4 v = ((const float4*)(feat + ((size_t)s * H + h) * 128))[lane];
    float* op = out + ((size_t)d * H + h) * 128 + lane * 4;
    atomicAdd(op + 0, v.x * alpha);
    atomicAdd(op + 1, v.y * alpha);
    atomicAdd(op + 2, v.z * alpha);
    atomicAdd(op + 3, v.w * alpha);
}

// ---------------------------------------------------------------------------
static inline int cdiv(int a, int b) { return (a + b - 1) / b; }

extern "C" void kernel_launch(void* const* d_in, const int* in_sizes, int /*n_in*/,
                              void* d_out, int /*out_size*/, void* d_ws, size_t /*ws_size*/,
                              hipStream_t stream) {
    const float* node = (const float*)d_in[0];
    const int*   src  = (const int*)  d_in[1];
    const int*   dst  = (const int*)  d_in[2];
    const float* fc1  = (const float*)d_in[3];
    const float* al1  = (const float*)d_in[4];
    const float* ar1  = (const float*)d_in[5];
    const float* res1 = (const float*)d_in[6];
    const float* b1   = (const float*)d_in[7];
    const float* fc2  = (const float*)d_in[8];
    const float* al2  = (const float*)d_in[9];
    const float* ar2  = (const float*)d_in[10];
    const float* res2 = (const float*)d_in[11];
    const float* b2   = (const float*)d_in[12];
    float* out = (float*)d_out;

    const int D = 128;
    const int N = in_sizes[0] / D;     // 50000
    const int E = in_sizes[1];         // 800000
    const int H1 = 3;

    // ---- workspace carve-out (floats)
    float* ws = (float*)d_ws;
    size_t off = 0;
    auto carve = [&](size_t nflt) { float* p = ws + off; off += (nflt + 63) & ~(size_t)63; return p; };

    float* feat1 = carve((size_t)N * 3 * D);   // [N,3,128] f32 -- reused by layer 2
    float* x1    = carve((size_t)N * 3 * D);   // residual+bias -> aggregated (f32)
    float* el1   = carve((size_t)N * H1);
    float* er1   = carve((size_t)N * H1);
    float* m1    = carve((size_t)N * H1);
    float* s1    = carve((size_t)N * H1);
    float* ex1   = carve((size_t)E * H1);
    unsigned short* node_bf = (unsigned short*)carve((size_t)N * D / 2);      // [N,128] bf16
    unsigned short* fc1_bf  = (unsigned short*)carve((size_t)384 * 128 / 2);
    unsigned short* res1_bf = (unsigned short*)carve((size_t)384 * 128 / 2);
    unsigned short* fc2_bf  = (unsigned short*)carve((size_t)128 * 384 / 2);
    unsigned short* res2_bf = (unsigned short*)carve((size_t)128 * 384 / 2);

    // layer-2 buffers alias the (dead after layer-1 aggregation) feat1 region:
    // feat2 [N*128 f32] + el2/er2/m2/s2 [N each] + ex2 [E] + x1_bf [N*384 bf16]
    float* feat2 = feat1;
    float* el2   = feat1 + (size_t)N * D;
    float* er2   = el2 + N;
    float* m2    = er2 + N;
    float* s2    = m2 + N;
    float* ex2   = s2 + N;
    unsigned short* x1_bf = (unsigned short*)(ex2 + E);   // [N,384] bf16

    const dim3 blk256(256), blk128(128);

    // ---- one-shot bf16 conversions of GEMM operands
    f32_to_bf16_kernel<<<cdiv(N * D / 8, 256), blk256, 0, stream>>>(node, node_bf, N * D);
    f32_to_bf16_kernel<<<cdiv(384 * 128 / 8, 256), blk256, 0, stream>>>(fc1,  fc1_bf,  384 * 128);
    f32_to_bf16_kernel<<<cdiv(384 * 128 / 8, 256), blk256, 0, stream>>>(res1, res1_bf, 384 * 128);
    f32_to_bf16_kernel<<<cdiv(128 * 384 / 8, 256), blk256, 0, stream>>>(fc2,  fc2_bf,  128 * 384);
    f32_to_bf16_kernel<<<cdiv(128 * 384 / 8, 256), blk256, 0, stream>>>(res2, res2_bf, 128 * 384);

    // ================= Layer 1 (in=128, out=128, heads=3) =================
    wmma_gemm_kernel<<<dim3(cdiv(N, 128), 6), blk256, 0, stream>>>(node_bf, fc1_bf,  nullptr, feat1, N, 128, 384);
    wmma_gemm_kernel<<<dim3(cdiv(N, 128), 6), blk256, 0, stream>>>(node_bf, res1_bf, b1,      x1,    N, 128, 384);
    attn_dot_kernel<3><<<cdiv(N * H1, 4), blk128, 0, stream>>>(feat1, al1, ar1, el1, er1, N * H1);
    init_ms_kernel<<<cdiv(N * H1, 256), blk256, 0, stream>>>(m1, s1, N * H1);
    edge_max_kernel<3><<<cdiv(E * H1, 256), blk256, 0, stream>>>(src, dst, el1, er1, ex1, m1, E * H1);
    edge_exp_kernel<3><<<cdiv(E * H1, 256), blk256, 0, stream>>>(dst, ex1, m1, s1, E * H1);
    edge_aggregate_kernel<3><<<cdiv(E * H1, 8), blk256, 0, stream>>>(src, dst, ex1, s1, feat1, x1, E * H1);
    // ELU + bf16 conversion -> x1_bf is the [N,384] layer-2 GEMM input
    elu_to_bf16_kernel<<<cdiv(N * 384 / 8, 256), blk256, 0, stream>>>(x1, x1_bf, N * 384);

    // ================= Layer 2 (in=384, out=128, heads=1) =================
    wmma_gemm_kernel<<<dim3(cdiv(N, 128), 2), blk256, 0, stream>>>(x1_bf, fc2_bf,  nullptr, feat2, N, 384, 128);
    wmma_gemm_kernel<<<dim3(cdiv(N, 128), 2), blk256, 0, stream>>>(x1_bf, res2_bf, b2,      out,   N, 384, 128);
    attn_dot_kernel<1><<<cdiv(N, 4), blk128, 0, stream>>>(feat2, al2, ar2, el2, er2, N);
    init_ms_kernel<<<cdiv(N, 256), blk256, 0, stream>>>(m2, s2, N);
    edge_max_kernel<1><<<cdiv(E, 256), blk256, 0, stream>>>(src, dst, el2, er2, ex2, m2, E);
    edge_exp_kernel<1><<<cdiv(E, 256), blk256, 0, stream>>>(dst, ex2, m2, s2, E);
    edge_aggregate_kernel<1><<<cdiv(E, 8), blk256, 0, stream>>>(src, dst, ex2, s2, feat2, out, E);
}
